// PDF_89000312308226
// MI455X (gfx1250) — matrix operand order repaired and checked by reading, not compiled, not run on hardware
//
#include <hip/hip_runtime.h>
#include <hip/hip_bf16.h>

// ---------------------------------------------------------------------------
// Problem constants (from reference): DIM=512, N_V=10000, N_F=20000
// d_in: 0 primal(10000x512 f32) 1 dual(20000x512 f32) 2 A(10000x20000 f32)
//       3 faces(20000x3 i64)    4 W_primal(512x1024)  5 b_primal(512)
//       6 W_dual(512x1024)      7 b_dual(512)
// d_out: out_primal (10000x512) then out_dual (20000x512), fp32
// ---------------------------------------------------------------------------

#if defined(__AMDGCN__) &&                                        \
    __has_builtin(__builtin_amdgcn_global_load_async_to_lds_b64) && \
    __has_builtin(__builtin_amdgcn_s_wait_asynccnt)
#define ASYNC_LDS 1
#else
#define ASYNC_LDS 0
#endif

typedef __attribute__((ext_vector_type(16))) __bf16 v16bf;
typedef __attribute__((ext_vector_type(16))) float  v16f;
typedef __attribute__((ext_vector_type(8)))  float  v8f;
typedef __attribute__((ext_vector_type(2)))  int    v2i;

#if ASYNC_LDS
typedef __attribute__((address_space(1))) v2i* gv2i_p;   // global (device) AS
typedef __attribute__((address_space(3))) v2i* lv2i_p;   // LDS (shared) AS
#endif

union FragBF {
    v16bf v;
    unsigned short s[16];
    unsigned int u[8];
    uint4 q[2];
};

// 16x fp32 -> 16x bf16 (RNE); compiler selects native packed cvt when present
__device__ __forceinline__ v16bf cvt16f_bf16(const float4 a0, const float4 a1,
                                             const float4 a2, const float4 a3) {
    v16f t;
    t[0]  = a0.x; t[1]  = a0.y; t[2]  = a0.z; t[3]  = a0.w;
    t[4]  = a1.x; t[5]  = a1.y; t[6]  = a1.z; t[7]  = a1.w;
    t[8]  = a2.x; t[9]  = a2.y; t[10] = a2.z; t[11] = a2.w;
    t[12] = a3.x; t[13] = a3.y; t[14] = a3.z; t[15] = a3.w;
    return __builtin_convertvector(t, v16bf);
}

// single fp32 -> bf16 bits
__device__ __forceinline__ unsigned short f2bf(float x) {
    union { __bf16 h; unsigned short s; } cv;
    cv.h = (__bf16)x;
    return cv.s;
}

// ---------------------------------------------------------------------------
// Kernel 1: f = mean_j |primal[faces[i,j]] - dual[i]|  (per face, per dim)
// Writes f (fp32, row-major 20000x512) and fT (bf16 bits, 512x20000) using an
// LDS transpose tile so fT stores are 16B-vector writes.
// Block = 256 threads (8 waves), 16 faces per block, grid = 1250.
// ---------------------------------------------------------------------------
__global__ void face_feat_kernel(const float* __restrict__ primal,
                                 const float* __restrict__ dual,
                                 const long long* __restrict__ faces,
                                 float* __restrict__ f_out,
                                 unsigned short* __restrict__ fT) {
    __shared__ unsigned short sT[512 * 16];   // [dim][face_local] bf16 bits

    const int f0   = blockIdx.x * 16;
    const int lane = threadIdx.x & 31;
    const int wave = threadIdx.x >> 5;

    for (int j = 0; j < 2; ++j) {
        const int fl = wave * 2 + j;          // 0..15
        const int fi = f0 + fl;
        const long long v0 = faces[(size_t)fi * 3 + 0];
        const long long v1 = faces[(size_t)fi * 3 + 1];
        const long long v2 = faces[(size_t)fi * 3 + 2];
        const float* p0 = primal + (size_t)v0 * 512;
        const float* p1 = primal + (size_t)v1 * 512;
        const float* p2 = primal + (size_t)v2 * 512;
        const float* dl = dual   + (size_t)fi * 512;
        float*       fo = f_out  + (size_t)fi * 512;
        for (int d = lane; d < 512; d += 32) {
            const float dv  = dl[d];
            const float val = (fabsf(p0[d] - dv) + fabsf(p1[d] - dv) +
                               fabsf(p2[d] - dv)) * (1.0f / 3.0f);
            fo[d] = val;
            sT[d * 16 + fl] = f2bf(val);
        }
    }
    __syncthreads();
    // Write fT rows: 16 bf16 (32B) per dim row, rows striped over threads.
    for (int row = threadIdx.x; row < 512; row += 256) {
        const uint4 a = *(const uint4*)&sT[row * 16 + 0];
        const uint4 b = *(const uint4*)&sT[row * 16 + 8];
        uint4* dst = (uint4*)(fT + (size_t)row * 20000 + f0);
        dst[0] = a;
        dst[1] = b;
    }
}

// ---------------------------------------------------------------------------
// WMMA GEMM:  out[m, n] = relu?( sum_k Acat[m,k] * B[k,n] + bias[n] )
//   Acat rows: A0 for k < k_split, A1 for k >= k_split (fp32, row-major).
//   B: either bf16 "BT" with B[k,n] at BT[n*ldB + k]   (B_BF16 = true)
//      or fp32 W with B[k,n] at W[n*ldB + k]           (B_BF16 = false)
//   N fixed at 512. Block = 256 threads = 8 waves; block covers one 16-row
//   M-strip and the full N (wave w owns N in [w*64, w*64+64), 4 C-tiles).
//   A 16x32 fp32 chunk double-buffered in LDS; staged via async global->LDS
//   copies (ASYNCcnt) when available, overlapping the next copy with WMMAs.
// ---------------------------------------------------------------------------
template <bool B_BF16, bool RELU_BIAS>
__global__ void gemm_wmma_kernel(const float* __restrict__ A0, int ldA0,
                                 const float* __restrict__ A1, int ldA1,
                                 int k_split,
                                 const void* __restrict__ Bp, int ldB,
                                 const float* __restrict__ bias,
                                 float* __restrict__ out, int K) {
    __shared__ float sAf[2][16 * 32];         // double-buffered fp32 A chunk

    const int m0   = blockIdx.x * 16;
    const int tid  = threadIdx.x;
    const int lane = tid & 31;
    const int wave = tid >> 5;

    // staging coords: each thread copies 2 consecutive fp32 of A (8 bytes)
    const int srow = tid >> 4;                // 0..15
    const int scol = (tid & 15) * 2;          // 0,2,..,30

    // fragment coords (this code's contiguous k = 16*(lane/16) + s in both A and B)
    const int mrow16 = lane & 15;             // A-frag M row / B-frag N row
    const int c0     = (lane < 16) ? 0 : 16;  // A-frag local-K chunk base
    const int kboff  = (lane < 16) ? 0 : 16;  // B-frag K offset

    const float* srowA0 = A0 + (size_t)(m0 + srow) * ldA0;
    const float* srowA1 = A1 + (size_t)(m0 + srow) * ldA1;

    auto stage = [&](int k0, int buf) {
        const int k = k0 + scol;
        const float* src = (k < k_split) ? (srowA0 + k)
                                         : (srowA1 + (k - k_split));
        float* dst = &sAf[buf][srow * 32 + scol];
#if ASYNC_LDS
        __builtin_amdgcn_global_load_async_to_lds_b64(
            (gv2i_p)(unsigned long long)src,
            (lv2i_p)(unsigned int)(unsigned long long)dst,
            0, 0);
#else
        const float2 a2 = *(const float2*)src;
        __builtin_prefetch(src + 32, 0, 0);   // next K chunk (streamed A)
        *(float2*)dst = a2;
#endif
    };

    v8f acc[4];
#pragma unroll
    for (int t = 0; t < 4; ++t)
        acc[t] = (v8f){0.f, 0.f, 0.f, 0.f, 0.f, 0.f, 0.f, 0.f};

    const unsigned short* BT = (const unsigned short*)Bp;
    const float*          Wf = (const float*)Bp;

    const int steps = K >> 5;                 // K is a multiple of 32
    stage(0, 0);
    for (int i = 0; i < steps; ++i) {
        const int cur = i & 1;
        const int k0  = i << 5;
#if ASYNC_LDS
        __builtin_amdgcn_s_wait_asynccnt(0);  // my share of buf[cur] landed
#endif
        __syncthreads();                      // block-wide: buf[cur] ready,
                                              // buf[cur^1] no longer read
        if (i + 1 < steps) stage(k0 + 32, cur ^ 1);

        // ---- A fragment: 16 fp32 from LDS -> bf16 (vector convert) ----
        const float* ab = &sAf[cur][mrow16 * 32];
        const float4 a0 = *(const float4*)(ab + c0);
        const float4 a1 = *(const float4*)(ab + c0 + 4);
        const float4 a2 = *(const float4*)(ab + c0 + 8);
        const float4 a3 = *(const float4*)(ab + c0 + 12);
        const v16bf afv = cvt16f_bf16(a0, a1, a2, a3);

        const int kb = k0 + kboff;
#pragma unroll
        for (int t = 0; t < 4; ++t) {
            const int n = wave * 64 + t * 16 + mrow16;
            v16bf bfv;
            if (B_BF16) {
                FragBF bf;
                const uint4* bp = (const uint4*)(BT + (size_t)n * ldB + kb);
                bf.q[0] = bp[0];
                bf.q[1] = bp[1];
                bfv = bf.v;
            } else {
                const float4* wp = (const float4*)(Wf + (size_t)n * ldB + kb);
                bfv = cvt16f_bf16(wp[0], wp[1], wp[2], wp[3]);
            }
            acc[t] = __builtin_amdgcn_wmma_f32_16x16x32_bf16(
                false, afv, false, bfv, (short)0, acc[t], false, false);
        }
    }

    // ---- epilogue: C layout -> global (lane%16 = N col, lane/16 = M half) ---
    const int mh = (lane >> 4) * 8;
#pragma unroll
    for (int t = 0; t < 4; ++t) {
        const int n = wave * 64 + t * 16 + mrow16;
        const float bv = RELU_BIAS ? bias[n] : 0.0f;
#pragma unroll
        for (int r = 0; r < 8; ++r) {
            float v = acc[t][r] + bv;
            if (RELU_BIAS) v = fmaxf(v, 0.0f);
            out[(size_t)(m0 + mh + r) * 512 + n] = v;
        }
    }
}

// ---------------------------------------------------------------------------
extern "C" void kernel_launch(void* const* d_in, const int* in_sizes, int n_in,
                              void* d_out, int out_size, void* d_ws, size_t ws_size,
                              hipStream_t stream) {
    const float*     primal  = (const float*)d_in[0];
    const float*     dual    = (const float*)d_in[1];
    const float*     A       = (const float*)d_in[2];
    const long long* faces   = (const long long*)d_in[3];
    const float*     W_prim  = (const float*)d_in[4];
    const float*     b_prim  = (const float*)d_in[5];
    const float*     W_dual  = (const float*)d_in[6];
    const float*     b_dual  = (const float*)d_in[7];

    float* out_primal = (float*)d_out;
    float* out_dual   = (float*)d_out + (size_t)10000 * 512;

    // workspace layout (256B aligned): f fp32 | fT bf16 | mapped fp32
    char* ws = (char*)d_ws;
    float*          f_f32  = (float*)ws;                           // 40,960,000 B
    unsigned short* fT     = (unsigned short*)(ws + 40960256);     // 20,480,000 B
    float*          mapped = (float*)(ws + 40960256 + 20480256);   // 20,480,000 B

    // 1) face features f (fp32) + transposed bf16 fT
    face_feat_kernel<<<1250, 256, 0, stream>>>(primal, dual, faces, f_f32, fT);

    // 2) mapped = A @ f        (M=10000, K=20000, N=512), B from bf16 fT
    gemm_wmma_kernel<true, false><<<625, 256, 0, stream>>>(
        A, 20000, A, 20000, /*k_split=*/20000, fT, 20000,
        nullptr, mapped, /*K=*/20000);

    // 3) out_dual = relu([dual | f] @ W_dual^T + b_dual)   (M=20000, K=1024)
    gemm_wmma_kernel<false, true><<<1250, 256, 0, stream>>>(
        dual, 512, f_f32, 512, /*k_split=*/512, W_dual, 1024,
        b_dual, out_dual, /*K=*/1024);

    // 4) out_primal = relu([primal | mapped] @ W_primal^T + b_primal)
    gemm_wmma_kernel<false, true><<<625, 256, 0, stream>>>(
        primal, 512, mapped, 512, /*k_split=*/512, W_prim, 1024,
        b_prim, out_primal, /*K=*/1024);
}
